// GCN_60370060312856
// MI455X (gfx1250) — compile-verified, matrix operand order checked by reading, not run on hardware
//
#include <hip/hip_runtime.h>
#include <hip/hip_bf16.h>
#include <stdint.h>

// ---------------- problem constants (from reference setup_inputs) ----------
#define N_NODES 50000
#define F_IN    1433
#define K_PAD   1440      // F_IN rounded to multiple of 32
#define KT1     45        // K_PAD / 32 chunks for layer-1 GEMM
#define KMAIN   44        // fully in-bounds chunks (44*32 = 1408)
#define K_TAIL_VALID 25   // 1433 - 1408 valid elements in tail chunk
#define HID     128
#define C_OUT   7
#define C_PAD   16
#define H2_STRIDE 8

typedef __attribute__((ext_vector_type(16))) __bf16 v16bf;
typedef __attribute__((ext_vector_type(8)))  float  v8f;

// native RNE conversion: lets clang emit v_cvt_pk_bf16_f32 on gfx1250
__device__ __forceinline__ __bf16 f2bf(float f) { return (__bf16)f; }

__device__ __forceinline__ void wait_async0() {
#if __has_builtin(__builtin_amdgcn_s_wait_asynccnt)
  __builtin_amdgcn_s_wait_asynccnt(0);
#else
  asm volatile("s_wait_asynccnt 0x0" ::: "memory");
#endif
}

// async copy one dword from global to an LDS byte offset (gfx1250 path)
__device__ __forceinline__ void async_g2l_b32(unsigned lds_byte, const float* g) {
  asm volatile("global_load_async_to_lds_b32 %0, %1, off"
               :: "v"(lds_byte), "v"(g) : "memory");
}

// ---------------- utility kernels -----------------------------------------
__global__ void zeroK(float* __restrict__ p, int n) {
  int i = blockIdx.x * blockDim.x + threadIdx.x;
  if (i < n) p[i] = 0.0f;
}

__global__ void degK(const long long* __restrict__ ei, float* __restrict__ deg, int E) {
  int e = blockIdx.x * blockDim.x + threadIdx.x;
  if (e < E) atomicAdd(&deg[(int)ei[(size_t)E + e]], 1.0f);
}

__global__ void dinvK(float* __restrict__ deg, int n) {
  int i = blockIdx.x * blockDim.x + threadIdx.x;
  if (i < n) deg[i] = rsqrtf(deg[i] + 1.0f);   // +1 self loop; deg -> dinv in place
}

// W1 -> bf16 in WMMA B-fragment order:
//   w1s[(((kt*8)+nt)*32 + lane)*16 + idx], K = kt*32 + koff + idx + (idx>=8 ? 8 : 0)
__global__ void convW1(const float* __restrict__ W1, __bf16* __restrict__ w1s) {
  int i = blockIdx.x * blockDim.x + threadIdx.x;   // KT1*8*32*16
  if (i >= KT1 * 8 * 512) return;
  int idx  = i & 15;
  int lane = (i >> 4) & 31;
  int nt   = (i >> 9) & 7;
  int kt   = i >> 12;
  int koff = (lane < 16) ? 0 : 8;
  int k = kt * 32 + koff + idx + ((idx >= 8) ? 8 : 0);
  int n = nt * 16 + (lane & 15);
  w1s[i] = (k < F_IN) ? f2bf(W1[(size_t)k * HID + n]) : f2bf(0.0f);
}

// W2 -> bf16 fragment order (single N tile, cols 7..15 zero)
__global__ void convW2(const float* __restrict__ W2, __bf16* __restrict__ w2s) {
  int i = blockIdx.x * blockDim.x + threadIdx.x;   // 4*32*16
  if (i >= 4 * 512) return;
  int idx  = i & 15;
  int lane = (i >> 4) & 31;
  int kt   = i >> 9;
  int koff = (lane < 16) ? 0 : 8;
  int k = kt * 32 + koff + idx + ((idx >= 8) ? 8 : 0);
  int n = lane & 15;
  w2s[i] = (n < C_OUT) ? f2bf(W2[(size_t)k * C_OUT + n]) : f2bf(0.0f);
}

// ---------------- GEMM1: h[50000,128] = x @ W1 (bf16 WMMA, async LDS) ------
__global__ __launch_bounds__(256) void gemm1K(const float* __restrict__ x,
                                              const __bf16* __restrict__ w1s,
                                              float* __restrict__ h) {
  __shared__ __align__(16) float As[512];          // [fragLane][16] f32
  const int tid  = threadIdx.x;
  const int wave = tid >> 5;
  const int lane = tid & 31;
  const int m0   = blockIdx.x * 16;

  // two elements per thread: e0 = tid, e1 = tid+256; element e = (row, k)
  const int e0 = tid, e1 = tid + 256;
  const int r0 = e0 >> 5, kk0 = e0 & 31;
  const int r1 = e1 >> 5, kk1 = e1 & 31;
  // fragment slot of (row,k): lane = row + 16*bit3(k), idx = (k&7) + (k>=16)*8
  const int fl0 = r0 + (((kk0 >> 3) & 1) << 4);
  const int fl1 = r1 + (((kk1 >> 3) & 1) << 4);
  const int fi0 = (kk0 & 7) + ((kk0 >= 16) ? 8 : 0);
  const int fi1 = (kk1 & 7) + ((kk1 >= 16) ? 8 : 0);
  const unsigned ldsBase = (unsigned)(size_t)(void*)As;
  const unsigned ldsD0 = ldsBase + (unsigned)(fl0 * 16 + fi0) * 4u;
  const unsigned ldsD1 = ldsBase + (unsigned)(fl1 * 16 + fi1) * 4u;
  const float* g0 = x + (size_t)(m0 + r0) * F_IN + kk0;
  const float* g1 = x + (size_t)(m0 + r1) * F_IN + kk1;

  // tail-chunk clamped pointers (always-valid load + select, no exec div.)
  const int kc0 = (kk0 < K_TAIL_VALID) ? kk0 : (K_TAIL_VALID - 1);
  const int kc1 = (kk1 < K_TAIL_VALID) ? kk1 : (K_TAIL_VALID - 1);
  const float* gt0 = x + (size_t)(m0 + r0) * F_IN + KMAIN * 32 + kc0;
  const float* gt1 = x + (size_t)(m0 + r1) * F_IN + KMAIN * 32 + kc1;

  v8f acc = {};
  for (int kt = 0; kt < KT1; ++kt) {
    __syncthreads();                               // previous consumers done
    if (kt < KMAIN) {
      async_g2l_b32(ldsD0, g0);
      async_g2l_b32(ldsD1, g1);
      wait_async0();                               // own async writes done
    } else {                                       // tail chunk, zero-padded
      float v0 = gt0[0], v1 = gt1[0];
      As[fl0 * 16 + fi0] = (kk0 < K_TAIL_VALID) ? v0 : 0.0f;
      As[fl1 * 16 + fi1] = (kk1 < K_TAIL_VALID) ? v1 : 0.0f;
    }
    __syncthreads();                               // all writes visible

    // A fragment: 16 contiguous f32 per lane (4x ds_load_b128) -> bf16
    const float4* Af = (const float4*)(As + lane * 16);
    v16bf a;
#pragma unroll
    for (int j = 0; j < 4; ++j) {
      float4 t = Af[j];
      a[4 * j + 0] = f2bf(t.x);
      a[4 * j + 1] = f2bf(t.y);
      a[4 * j + 2] = f2bf(t.z);
      a[4 * j + 3] = f2bf(t.w);
    }
    // B fragment: one contiguous 32B load per lane
    v16bf b = *(const v16bf*)(w1s + ((size_t)((kt * 8 + wave) * 32 + lane) * 16));

    acc = __builtin_amdgcn_wmma_f32_16x16x32_bf16(false, a, false, b,
                                                  (short)0, acc, false, false);
    g0 += 32;
    g1 += 32;
  }

  const int col   = lane & 15;
  const int mbase = (lane < 16) ? 0 : 8;
#pragma unroll
  for (int r = 0; r < 8; ++r)
    h[(size_t)(m0 + mbase + r) * HID + wave * 16 + col] = acc[r];
}

// ---------------- edge aggregation, layer 1 (wave per edge, float4/lane) ---
__global__ __launch_bounds__(256) void agg1K(const long long* __restrict__ ei,
                                             const float* __restrict__ dinv,
                                             const float* __restrict__ h,
                                             float* __restrict__ hagg, int E) {
  int e = blockIdx.x * (blockDim.x >> 5) + (threadIdx.x >> 5);
  if (e >= E) return;
  int lane = threadIdx.x & 31;
  int s = (int)ei[e];
  int d = (int)ei[(size_t)E + e];
  float nm = dinv[s] * dinv[d];
  const float4* hs = (const float4*)(h + (size_t)s * HID);
  float4 v = hs[lane];
  float* od = hagg + (size_t)d * HID + lane * 4;
  atomicAdd(od + 0, nm * v.x);
  atomicAdd(od + 1, nm * v.y);
  atomicAdd(od + 2, nm * v.z);
  atomicAdd(od + 3, nm * v.w);
}

// ---------------- GEMM2: h2 = relu(hagg + dinv^2*h + b1) @ W2 (bf16 WMMA) --
__global__ __launch_bounds__(256) void gemm2K(const float* __restrict__ hagg,
                                              const float* __restrict__ h,
                                              const float* __restrict__ dinv,
                                              const float* __restrict__ b1,
                                              const __bf16* __restrict__ w2s,
                                              float* __restrict__ h2) {
  const int tid  = threadIdx.x;
  const int wave = tid >> 5;
  const int lane = tid & 31;
  const int mtile = blockIdx.x * 8 + wave;
  if (mtile * 16 >= N_NODES) return;               // wave-uniform
  const int m0   = mtile * 16;
  const int row  = lane & 15;
  const int koff = (lane < 16) ? 0 : 8;

  const float di = dinv[m0 + row];
  const float d2 = di * di;
  const float* ra = hagg + (size_t)(m0 + row) * HID;
  const float* rh = h    + (size_t)(m0 + row) * HID;

  v8f acc = {};
#pragma unroll
  for (int kt = 0; kt < 4; ++kt) {
    const int ka = kt * 32 + koff;        // 8 contiguous K for low half
    const int kb = ka + 16;               // 8 contiguous K for high half
    float4 a0 = *(const float4*)(ra + ka);
    float4 a1 = *(const float4*)(ra + ka + 4);
    float4 a2 = *(const float4*)(ra + kb);
    float4 a3 = *(const float4*)(ra + kb + 4);
    float4 h0 = *(const float4*)(rh + ka);
    float4 h1 = *(const float4*)(rh + ka + 4);
    float4 h2v = *(const float4*)(rh + kb);
    float4 h3 = *(const float4*)(rh + kb + 4);
    float4 c0 = *(const float4*)(b1 + ka);
    float4 c1 = *(const float4*)(b1 + ka + 4);
    float4 c2 = *(const float4*)(b1 + kb);
    float4 c3 = *(const float4*)(b1 + kb + 4);

    float va[16];
    va[0] = a0.x + d2 * h0.x + c0.x;  va[1] = a0.y + d2 * h0.y + c0.y;
    va[2] = a0.z + d2 * h0.z + c0.z;  va[3] = a0.w + d2 * h0.w + c0.w;
    va[4] = a1.x + d2 * h1.x + c1.x;  va[5] = a1.y + d2 * h1.y + c1.y;
    va[6] = a1.z + d2 * h1.z + c1.z;  va[7] = a1.w + d2 * h1.w + c1.w;
    va[8]  = a2.x + d2 * h2v.x + c2.x; va[9]  = a2.y + d2 * h2v.y + c2.y;
    va[10] = a2.z + d2 * h2v.z + c2.z; va[11] = a2.w + d2 * h2v.w + c2.w;
    va[12] = a3.x + d2 * h3.x + c3.x;  va[13] = a3.y + d2 * h3.y + c3.y;
    va[14] = a3.z + d2 * h3.z + c3.z;  va[15] = a3.w + d2 * h3.w + c3.w;

    v16bf a;
#pragma unroll
    for (int i = 0; i < 16; ++i) a[i] = f2bf(va[i] > 0.0f ? va[i] : 0.0f);

    v16bf b = *(const v16bf*)(w2s + ((size_t)(kt * 32 + lane) * 16));
    acc = __builtin_amdgcn_wmma_f32_16x16x32_bf16(false, a, false, b,
                                                  (short)0, acc, false, false);
  }

  const int mbase = (lane < 16) ? 0 : 8;
  if (row < C_OUT) {
#pragma unroll
    for (int r = 0; r < 8; ++r)
      h2[(size_t)(m0 + mbase + r) * H2_STRIDE + row] = acc[r];
  }
}

// ---------------- edge aggregation, layer 2 (thread per edge, 7 feats) -----
__global__ void agg2K(const long long* __restrict__ ei,
                      const float* __restrict__ dinv,
                      const float* __restrict__ h2,
                      float* __restrict__ out, int E) {
  int e = blockIdx.x * blockDim.x + threadIdx.x;
  if (e >= E) return;
  int s = (int)ei[e];
  int d = (int)ei[(size_t)E + e];
  float nm = dinv[s] * dinv[d];
  const float4* hs = (const float4*)(h2 + (size_t)s * H2_STRIDE);
  float4 v0 = hs[0], v1 = hs[1];                    // 8 floats (7 valid + pad)
  float* od = out + (size_t)d * C_OUT;
  atomicAdd(od + 0, nm * v0.x);
  atomicAdd(od + 1, nm * v0.y);
  atomicAdd(od + 2, nm * v0.z);
  atomicAdd(od + 3, nm * v0.w);
  atomicAdd(od + 4, nm * v1.x);
  atomicAdd(od + 5, nm * v1.y);
  atomicAdd(od + 6, nm * v1.z);
}

// ---------------- final: out += dinv^2*h2 + b2 -----------------------------
__global__ void finalK(const float* __restrict__ dinv,
                       const float* __restrict__ h2,
                       const float* __restrict__ b2,
                       float* __restrict__ out, int total) {
  int i = blockIdx.x * blockDim.x + threadIdx.x;
  if (i >= total) return;
  int node = i / C_OUT, c = i - node * C_OUT;
  float di = dinv[node];
  out[i] += di * di * h2[(size_t)node * H2_STRIDE + c] + b2[c];
}

// ---------------- host-side launch -----------------------------------------
static inline char* carve(char*& p, size_t bytes) {
  char* r = p;
  p += (bytes + 255) & ~size_t(255);
  return r;
}

extern "C" void kernel_launch(void* const* d_in, const int* in_sizes, int n_in,
                              void* d_out, int out_size, void* d_ws, size_t ws_size,
                              hipStream_t stream) {
  const float*     x  = (const float*)d_in[0];
  const long long* ei = (const long long*)d_in[1];   // int64 in reference
  const float*     W1 = (const float*)d_in[2];
  const float*     b1 = (const float*)d_in[3];
  const float*     W2 = (const float*)d_in[4];
  const float*     b2 = (const float*)d_in[5];
  float* out = (float*)d_out;

  const int E = in_sizes[1] / 2;                     // 1,600,000

  char* ws = (char*)d_ws;
  float*  dinv = (float*)carve(ws, (size_t)N_NODES * 4);
  __bf16* w1s  = (__bf16*)carve(ws, (size_t)KT1 * 8 * 512 * 2);
  __bf16* w2s  = (__bf16*)carve(ws, (size_t)4 * 512 * 2);
  float*  h    = (float*)carve(ws, (size_t)N_NODES * HID * 4);
  float*  hagg = (float*)carve(ws, (size_t)N_NODES * HID * 4);
  float*  h2   = (float*)carve(ws, (size_t)N_NODES * H2_STRIDE * 4);

  // re-zero accumulation targets every call (deterministic, capture-safe)
  zeroK<<<(N_NODES + 255) / 256, 256, 0, stream>>>(dinv, N_NODES);
  zeroK<<<(N_NODES * HID + 255) / 256, 256, 0, stream>>>(hagg, N_NODES * HID);
  zeroK<<<(out_size + 255) / 256, 256, 0, stream>>>(out, out_size);

  degK<<<(E + 255) / 256, 256, 0, stream>>>(ei, dinv, E);
  dinvK<<<(N_NODES + 255) / 256, 256, 0, stream>>>(dinv, N_NODES);

  convW1<<<(KT1 * 8 * 512 + 255) / 256, 256, 0, stream>>>(W1, w1s);
  convW2<<<(4 * 512 + 255) / 256, 256, 0, stream>>>(W2, w2s);

  gemm1K<<<N_NODES / 16, 256, 0, stream>>>(x, w1s, h);
  agg1K<<<(E + 7) / 8, 256, 0, stream>>>(ei, dinv, h, hagg, E);

  gemm2K<<<(N_NODES / 16 + 7) / 8, 256, 0, stream>>>(hagg, h, dinv, b1, w2s, h2);
  agg2K<<<(E + 255) / 256, 256, 0, stream>>>(ei, dinv, h2, out, E);

  finalK<<<(N_NODES * C_OUT + 255) / 256, 256, 0, stream>>>(dinv, h2, b2, out,
                                                            N_NODES * C_OUT);
}
